// MultiHeadAttention_33621003993637
// MI455X (gfx1250) — matrix operand order repaired and checked by reading, not compiled, not run on hardware
//
#include <hip/hip_runtime.h>
#include <math.h>

// ---------------------------------------------------------------------------
// MHA (B=4, S=2048, D=512, H=8, d_k=64) for gfx1250 (MI455X), bf16 WMMA.
// Stage 1: fp32 -> bf16 conversion (vectorized, pure bandwidth)
// Stage 2: Q/K/V projections, 32x32 register-blocked bf16 WMMA GEMM
//          (Q pre-scaled by 1/sqrt(dk), V stored transposed per head)
// Stage 3: fused flash attention, LDS-staged double-buffered K/V blocks
//          (async global->LDS copies when the toolchain exposes them)
// Stage 4: output projection, 32x32 register-blocked, fp32 result + bias
// ---------------------------------------------------------------------------

#define BATCH 4
#define SEQ   2048
#define DMODEL 512
#define NHEAD 8
#define DK    64
#define LOG2E 1.44269504088896340736f

typedef __attribute__((ext_vector_type(16))) __bf16 v16bf;
typedef __attribute__((ext_vector_type(8)))  float  v8f;
typedef __attribute__((ext_vector_type(4)))  int    v4i_;

#if defined(__HIP_DEVICE_COMPILE__) && __has_builtin(__builtin_amdgcn_global_load_async_to_lds_b128)
#define HAVE_ASYNC_LDS 1
#else
#define HAVE_ASYNC_LDS 0
#endif

#if defined(__HIP_DEVICE_COMPILE__) && __has_builtin(__builtin_amdgcn_s_wait_asynccnt)
#define WAIT_ASYNC() __builtin_amdgcn_s_wait_asynccnt(0)
#elif defined(__HIP_DEVICE_COMPILE__)
#define WAIT_ASYNC() asm volatile("s_wait_asynccnt 0" ::: "memory")
#else
#define WAIT_ASYNC()
#endif

union FragU {
  v16bf bf;
  float4 f4[2];
};

// A-operand (16x32 bf16, row-major source, leading dim ld):
// lane half = lane>>4, row = lane&15; element j -> k = (j>>3)*16 + half*8 + (j&7)
// => two contiguous 16B chunks at k = half*8 and k = 16 + half*8.
template <typename P>
__device__ inline v16bf load_frag_a(const P* base, int ld, int lane) {
  const int half = lane >> 4;
  const int row  = lane & 15;
  const __bf16* p = (const __bf16*)base + (size_t)row * ld + half * 8;
  FragU u;
  u.f4[0] = *(const float4*)(p);        // k = half*8 .. half*8+7
  u.f4[1] = *(const float4*)(p + 16);   // k = 16+half*8 .. 16+half*8+7
  return u.bf;
}

// B-operand (32x16 bf16). Source stored as rows of N with K contiguous
// (i.e. base[n*ld + k]). n = lane&15; element j -> k = (j>>3)*16 + half*8 + (j&7)
// => two contiguous 16B chunks at k = half*8 and k = 16 + half*8 (matches A).
template <typename P>
__device__ inline v16bf load_frag_b(const P* base, int ld, int lane) {
  const int half = lane >> 4;
  const int n    = lane & 15;
  const __bf16* p = (const __bf16*)base + (size_t)n * ld + half * 8;
  FragU u;
  u.f4[0] = *(const float4*)(p);
  u.f4[1] = *(const float4*)(p + 16);
  return u.bf;
}

__device__ inline v8f wmma_bf16(v16bf a, v16bf b, v8f c) {
  return __builtin_amdgcn_wmma_f32_16x16x32_bf16(
      /*neg_a=*/false, a, /*neg_b=*/false, b,
      /*c_mod=*/(short)0, c, /*reuse_a=*/false, /*reuse_b=*/false);
}

#if HAVE_ASYNC_LDS
// 16-byte (8 x bf16) async global -> LDS copy, tracked with ASYNCcnt.
// Signature (from clang diagnostic): param0 = global int4*, 4 args total.
__device__ inline void async_copy16(__bf16* ldst, const __bf16* gsrc) {
  __builtin_amdgcn_global_load_async_to_lds_b128(
      (__attribute__((address_space(1))) v4i_*)(gsrc),
      (__attribute__((address_space(3))) v4i_*)(ldst),
      /*offset=*/0, /*cpol=*/0);
}
#endif

// ---------------------------------------------------------------------------
// Stage 1: fp32 -> bf16 conversion, 8 elements / thread (b128 in, b128 out)
// ---------------------------------------------------------------------------
union Bf8U { float4 f4; __bf16 b[8]; };

__global__ void cvt_f32_bf16_v8(const float* __restrict__ src,
                                __bf16* __restrict__ dst, int n8) {
  int i = blockIdx.x * blockDim.x + threadIdx.x;
  if (i < n8) {
    const float4* s = (const float4*)src;
    float4 a = s[2 * i];
    float4 c = s[2 * i + 1];
    Bf8U u;
    u.b[0] = (__bf16)a.x; u.b[1] = (__bf16)a.y;
    u.b[2] = (__bf16)a.z; u.b[3] = (__bf16)a.w;
    u.b[4] = (__bf16)c.x; u.b[5] = (__bf16)c.y;
    u.b[6] = (__bf16)c.z; u.b[7] = (__bf16)c.w;
    ((float4*)dst)[i] = u.f4;
  }
}

// ---------------------------------------------------------------------------
// Stage 2: projection GEMM, 32x32 C tile per wave (2x2 WMMA register block).
// out[m][e] = sum_d X[m][d] * W[e][d] + bias[e]
// mode 0: head-split   out[(b*H+h)][s][dk]   (Q scale=1/sqrt(dk), K scale=1)
// mode 1: transposed   out[(b*H+h)][dk][s]   (V)
// Block = 8 waves (2 M x 4 N) = 64 rows x 128 cols.
// ---------------------------------------------------------------------------
__global__ __launch_bounds__(256) void proj_bf16_kernel(
    const __bf16* __restrict__ X,    // [8192][512]
    const __bf16* __restrict__ W,    // [512][512]  (rows = output dim e)
    const float*  __restrict__ bias, // [512]
    __bf16* __restrict__ out,
    int mode, float scale) {
  const int lane  = threadIdx.x & 31;
  const int wave  = threadIdx.x >> 5;
  const int Nbase = blockIdx.x * 128 + (wave & 3) * 32;
  const int Mbase = blockIdx.y * 64 + (wave >> 2) * 32;

  v8f c[2][2] = {};
  for (int kk = 0; kk < DMODEL; kk += 32) {
    v16bf a0 = load_frag_a(X + (size_t)Mbase * DMODEL + kk,        DMODEL, lane);
    v16bf a1 = load_frag_a(X + (size_t)(Mbase + 16) * DMODEL + kk, DMODEL, lane);
    v16bf b0 = load_frag_b(W + (size_t)Nbase * DMODEL + kk,        DMODEL, lane);
    v16bf b1 = load_frag_b(W + (size_t)(Nbase + 16) * DMODEL + kk, DMODEL, lane);
    c[0][0] = wmma_bf16(a0, b0, c[0][0]);
    c[0][1] = wmma_bf16(a0, b1, c[0][1]);
    c[1][0] = wmma_bf16(a1, b0, c[1][0]);
    c[1][1] = wmma_bf16(a1, b1, c[1][1]);
  }

  const int half = lane >> 4;
#pragma unroll
  for (int ty = 0; ty < 2; ++ty) {
#pragma unroll
    for (int tx = 0; tx < 2; ++tx) {
      const int n  = Nbase + tx * 16 + (lane & 15);
      const int hh = n >> 6;
      const int dk = n & 63;
      const float bv = bias[n];
#pragma unroll
      for (int r = 0; r < 8; ++r) {
        const int m  = Mbase + ty * 16 + r + 8 * half;
        const int bb = m >> 11;
        const int s  = m & 2047;
        const float v = (c[ty][tx][r] + bv) * scale;
        size_t idx;
        if (mode == 0)
          idx = (((size_t)(bb * NHEAD + hh)) * SEQ + s) * DK + dk;
        else
          idx = (((size_t)(bb * NHEAD + hh)) * DK + dk) * SEQ + s;
        out[idx] = (__bf16)v;
      }
    }
  }
}

// ---------------------------------------------------------------------------
// Stage 3: fused flash attention with LDS-staged, double-buffered K/V.
// Grid: (B*H, SEQ/128). Block: 256 threads = 8 waves; each wave owns 16 q rows.
// Per 64-key block: cooperative 8KB K + 8KB V stage into LDS (async if
// available), 8 WMMAs for S = Q*K^T, online softmax, P via per-wave LDS tile,
// 8 WMMAs for O += P*V.  LDS: 2*8KB (K) + 2*8KB (V) + 16KB (P) = 48KB.
// ---------------------------------------------------------------------------
__global__ __launch_bounds__(256) void flash_attn_kernel(
    const __bf16* __restrict__ Q,   // [B*H][S][64], pre-scaled by 1/sqrt(dk)
    const __bf16* __restrict__ K,   // [B*H][S][64]
    const __bf16* __restrict__ Vt,  // [B*H][64][S]
    __bf16* __restrict__ O) {       // [B][S][512]
  __shared__ __bf16 kbuf[2][64 * DK];   // [key][dk]
  __shared__ __bf16 vbuf[2][DK * 64];   // [dk][key]
  __shared__ __bf16 pbuf[8][16 * 64];   // per-wave P tile

  const int t     = threadIdx.x;
  const int lane  = t & 31;
  const int wave  = t >> 5;
  const int half  = lane >> 4;
  const int n     = lane & 15;
  const int bh    = blockIdx.x;
  const int b     = bh >> 3;
  const int h     = bh & 7;
  const int qbase = blockIdx.y * 128 + wave * 16;

  const __bf16* Qh = Q  + (size_t)bh * SEQ * DK;
  const __bf16* Kh = K  + (size_t)bh * SEQ * DK;
  const __bf16* Vh = Vt + (size_t)bh * DK * SEQ;
  __bf16* pl = &pbuf[wave][0];

  // Q fragments for this wave's 16 rows (d_k = 64 -> two K=32 chunks)
  const v16bf qa0 = load_frag_a(Qh + (size_t)qbase * DK,      DK, lane);
  const v16bf qa1 = load_frag_a(Qh + (size_t)qbase * DK + 32, DK, lane);

  // staging indices: K block is 8KB contiguous; V block = 64 rows stride SEQ
  const int vrow = t >> 2;          // 0..63  (dk row)
  const int voff = (t & 3) * 16;    // 0/16/32/48 within key dim

  // ---- stage block 0 ----
#if HAVE_ASYNC_LDS
  {
    const __bf16* ks = Kh + t * 16;
    async_copy16(kbuf[0] + t * 16,     ks);
    async_copy16(kbuf[0] + t * 16 + 8, ks + 8);
    const __bf16* vs = Vh + (size_t)vrow * SEQ + voff;
    async_copy16(&vbuf[0][vrow * 64 + voff],     vs);
    async_copy16(&vbuf[0][vrow * 64 + voff + 8], vs + 8);
  }
  WAIT_ASYNC();
#else
  {
    const float4* ks = (const float4*)Kh;
    ((float4*)kbuf[0])[2 * t]     = ks[2 * t];
    ((float4*)kbuf[0])[2 * t + 1] = ks[2 * t + 1];
    const __bf16* vs = Vh + (size_t)vrow * SEQ + voff;
    *(float4*)(&vbuf[0][vrow * 64 + voff])     = *(const float4*)(vs);
    *(float4*)(&vbuf[0][vrow * 64 + voff + 8]) = *(const float4*)(vs + 8);
  }
#endif
  __syncthreads();

  v8f o0{}, o1{}, o2{}, o3{};
  float mrow[8], lrow[8];
#pragma unroll
  for (int r = 0; r < 8; ++r) { mrow[r] = -3.0e38f; lrow[r] = 0.0f; }

  for (int kb = 0; kb < SEQ; kb += 64) {
    const int cur = (kb >> 6) & 1;
    const int nxt = cur ^ 1;
    const int kn  = kb + 64;
    const bool havenext = (kn < SEQ);

#if HAVE_ASYNC_LDS
    // ---- issue async staging of next K/V block (no VGPRs consumed) ----
    if (havenext) {
      const __bf16* ks = Kh + (size_t)kn * DK + t * 16;
      async_copy16(kbuf[nxt] + t * 16,     ks);
      async_copy16(kbuf[nxt] + t * 16 + 8, ks + 8);
      const __bf16* vs = Vh + (size_t)vrow * SEQ + kn + voff;
      async_copy16(&vbuf[nxt][vrow * 64 + voff],     vs);
      async_copy16(&vbuf[nxt][vrow * 64 + voff + 8], vs + 8);
    }
#else
    // ---- prefetch next K/V block into registers (hide global latency) ----
    float4 pk0{}, pk1{}, pv0{}, pv1{};
    if (havenext) {
      const float4* ks = (const float4*)(Kh + (size_t)kn * DK);
      pk0 = ks[2 * t];
      pk1 = ks[2 * t + 1];
      const __bf16* vs = Vh + (size_t)vrow * SEQ + kn + voff;
      pv0 = *(const float4*)(vs);
      pv1 = *(const float4*)(vs + 8);
    }
#endif

    const __bf16* kt = kbuf[cur];
    const __bf16* vt = vbuf[cur];

    // ---- scores: 4 n-tiles of 16 keys, contraction over d_k = 64 ----
    v8f s0{}, s1{}, s2{}, s3{};
    {
      v16bf b0 = load_frag_b(kt +  0 * 64,      64, lane);
      v16bf b1 = load_frag_b(kt +  0 * 64 + 32, 64, lane);
      s0 = wmma_bf16(qa0, b0, s0); s0 = wmma_bf16(qa1, b1, s0);
      b0 = load_frag_b(kt + 16 * 64,      64, lane);
      b1 = load_frag_b(kt + 16 * 64 + 32, 64, lane);
      s1 = wmma_bf16(qa0, b0, s1); s1 = wmma_bf16(qa1, b1, s1);
      b0 = load_frag_b(kt + 32 * 64,      64, lane);
      b1 = load_frag_b(kt + 32 * 64 + 32, 64, lane);
      s2 = wmma_bf16(qa0, b0, s2); s2 = wmma_bf16(qa1, b1, s2);
      b0 = load_frag_b(kt + 48 * 64,      64, lane);
      b1 = load_frag_b(kt + 48 * 64 + 32, 64, lane);
      s3 = wmma_bf16(qa0, b0, s3); s3 = wmma_bf16(qa1, b1, s3);
    }

    // ---- online softmax: row stats live in 16-lane halves (C layout) ----
    float newm[8], sc[8];
#pragma unroll
    for (int r = 0; r < 8; ++r) {
      float pm = fmaxf(fmaxf(s0[r], s1[r]), fmaxf(s2[r], s3[r]));
      pm = fmaxf(pm, __shfl_xor(pm, 1, 32));
      pm = fmaxf(pm, __shfl_xor(pm, 2, 32));
      pm = fmaxf(pm, __shfl_xor(pm, 4, 32));
      pm = fmaxf(pm, __shfl_xor(pm, 8, 32));
      newm[r] = fmaxf(mrow[r], pm);
      sc[r]   = exp2f((mrow[r] - newm[r]) * LOG2E);
      mrow[r] = newm[r];
    }
#pragma unroll
    for (int r = 0; r < 8; ++r) {
      s0[r] = exp2f((s0[r] - newm[r]) * LOG2E);
      s1[r] = exp2f((s1[r] - newm[r]) * LOG2E);
      s2[r] = exp2f((s2[r] - newm[r]) * LOG2E);
      s3[r] = exp2f((s3[r] - newm[r]) * LOG2E);
      float ps = s0[r] + s1[r] + s2[r] + s3[r];
      ps += __shfl_xor(ps, 1, 32);
      ps += __shfl_xor(ps, 2, 32);
      ps += __shfl_xor(ps, 4, 32);
      ps += __shfl_xor(ps, 8, 32);
      lrow[r] = lrow[r] * sc[r] + ps;
      o0[r] *= sc[r]; o1[r] *= sc[r]; o2[r] *= sc[r]; o3[r] *= sc[r];
    }

    // ---- P: C-layout -> A-layout via per-wave LDS tile (16 x 64 bf16) ----
#pragma unroll
    for (int r = 0; r < 8; ++r) {
      const int row = r + 8 * half;
      pl[row * 64 +  0 + n] = (__bf16)s0[r];
      pl[row * 64 + 16 + n] = (__bf16)s1[r];
      pl[row * 64 + 32 + n] = (__bf16)s2[r];
      pl[row * 64 + 48 + n] = (__bf16)s3[r];
    }
    asm volatile("s_wait_dscnt 0" ::: "memory");
    const v16bf pa0 = load_frag_a(pl,      64, lane);  // keys kb+0..31
    const v16bf pa1 = load_frag_a(pl + 32, 64, lane);  // keys kb+32..63

    // ---- O += P * V  (vbuf is [dk][key], so B-operand loads are contiguous) ----
    {
      v16bf v0 = load_frag_b(vt +  0 * 64,      64, lane);
      v16bf v1 = load_frag_b(vt +  0 * 64 + 32, 64, lane);
      o0 = wmma_bf16(pa0, v0, o0); o0 = wmma_bf16(pa1, v1, o0);
      v0 = load_frag_b(vt + 16 * 64,      64, lane);
      v1 = load_frag_b(vt + 16 * 64 + 32, 64, lane);
      o1 = wmma_bf16(pa0, v0, o1); o1 = wmma_bf16(pa1, v1, o1);
      v0 = load_frag_b(vt + 32 * 64,      64, lane);
      v1 = load_frag_b(vt + 32 * 64 + 32, 64, lane);
      o2 = wmma_bf16(pa0, v0, o2); o2 = wmma_bf16(pa1, v1, o2);
      v0 = load_frag_b(vt + 48 * 64,      64, lane);
      v1 = load_frag_b(vt + 48 * 64 + 32, 64, lane);
      o3 = wmma_bf16(pa0, v0, o3); o3 = wmma_bf16(pa1, v1, o3);
    }

    // ---- finish staging of next block; one barrier per iteration ----
#if HAVE_ASYNC_LDS
    WAIT_ASYNC();
#else
    if (havenext) {
      ((float4*)kbuf[nxt])[2 * t]     = pk0;
      ((float4*)kbuf[nxt])[2 * t + 1] = pk1;
      *(float4*)(&vbuf[nxt][vrow * 64 + voff])     = pv0;
      *(float4*)(&vbuf[nxt][vrow * 64 + voff + 8]) = pv1;
    }
#endif
    __syncthreads();
  }

  // ---- normalize and write O as [B][S][512] (head-concatenated) ----
#pragma unroll
  for (int r = 0; r < 8; ++r) {
    const float inv = 1.0f / lrow[r];
    const int srow = qbase + r + 8 * half;
    __bf16* orow = O + ((size_t)b * SEQ + srow) * DMODEL + h * DK;
    orow[ 0 + n] = (__bf16)(o0[r] * inv);
    orow[16 + n] = (__bf16)(o1[r] * inv);
    orow[32 + n] = (__bf16)(o2[r] * inv);
    orow[48 + n] = (__bf16)(o3[r] * inv);
  }
}

// ---------------------------------------------------------------------------
// Stage 4: output projection, 32x32 register-blocked, fp32 result:
// out[m][e] = sum_d O[m][d] * Wo[e][d] + bo[e]
// ---------------------------------------------------------------------------
__global__ __launch_bounds__(256) void proj_out_kernel(
    const __bf16* __restrict__ X,    // [8192][512] bf16
    const __bf16* __restrict__ W,    // [512][512]  bf16
    const float*  __restrict__ bias, // [512]
    float* __restrict__ out) {       // [8192][512] fp32
  const int lane  = threadIdx.x & 31;
  const int wave  = threadIdx.x >> 5;
  const int Nbase = blockIdx.x * 128 + (wave & 3) * 32;
  const int Mbase = blockIdx.y * 64 + (wave >> 2) * 32;

  v8f c[2][2] = {};
  for (int kk = 0; kk < DMODEL; kk += 32) {
    v16bf a0 = load_frag_a(X + (size_t)Mbase * DMODEL + kk,        DMODEL, lane);
    v16bf a1 = load_frag_a(X + (size_t)(Mbase + 16) * DMODEL + kk, DMODEL, lane);
    v16bf b0 = load_frag_b(W + (size_t)Nbase * DMODEL + kk,        DMODEL, lane);
    v16bf b1 = load_frag_b(W + (size_t)(Nbase + 16) * DMODEL + kk, DMODEL, lane);
    c[0][0] = wmma_bf16(a0, b0, c[0][0]);
    c[0][1] = wmma_bf16(a0, b1, c[0][1]);
    c[1][0] = wmma_bf16(a1, b0, c[1][0]);
    c[1][1] = wmma_bf16(a1, b1, c[1][1]);
  }

  const int half = lane >> 4;
#pragma unroll
  for (int ty = 0; ty < 2; ++ty) {
#pragma unroll
    for (int tx = 0; tx < 2; ++tx) {
      const int n = Nbase + tx * 16 + (lane & 15);
      const float bv = bias[n];
#pragma unroll
      for (int r = 0; r < 8; ++r) {
        const int m = Mbase + ty * 16 + r + 8 * half;
        out[(size_t)m * DMODEL + n] = c[ty][tx][r] + bv;
      }
    }
  }
}

// ---------------------------------------------------------------------------
// Host-side orchestration
// ---------------------------------------------------------------------------
extern "C" void kernel_launch(void* const* d_in, const int* in_sizes, int n_in,
                              void* d_out, int out_size, void* d_ws, size_t ws_size,
                              hipStream_t stream) {
  const float* q_in = (const float*)d_in[0];
  const float* k_in = (const float*)d_in[1];
  const float* v_in = (const float*)d_in[2];
  const float* Wq = (const float*)d_in[3];
  const float* bq = (const float*)d_in[4];
  const float* Wk = (const float*)d_in[5];
  const float* bk = (const float*)d_in[6];
  const float* Wv = (const float*)d_in[7];
  const float* bv = (const float*)d_in[8];
  const float* Wo = (const float*)d_in[9];
  const float* bo = (const float*)d_in[10];
  float* out = (float*)d_out;

  const size_t nX = (size_t)BATCH * SEQ * DMODEL;   // 4,194,304
  const size_t nW = (size_t)DMODEL * DMODEL;        //   262,144

  // Workspace layout (bf16 elements), 256B-aligned chunks.
  char* ws = (char*)d_ws;
  size_t off = 0;
  auto take = [&](size_t elems) {
    __bf16* p = (__bf16*)(ws + off);
    off = (off + elems * sizeof(__bf16) + 255) & ~(size_t)255;
    return p;
  };
  __bf16* Xq  = take(nX);
  __bf16* Xk  = take(nX);
  __bf16* Xv  = take(nX);
  __bf16* Wqb = take(nW);
  __bf16* Wkb = take(nW);
  __bf16* Wvb = take(nW);
  __bf16* Wob = take(nW);
  __bf16* Qb  = take(nX);   // [B*H][S][64]
  __bf16* Kb  = take(nX);   // [B*H][S][64]
  __bf16* Vtb = take(nX);   // [B*H][64][S]
  __bf16* Ob  = take(nX);   // [B][S][512]
  (void)ws_size;

  const float scale = 1.0f / (sqrtf((float)DK) + 1e-8f);

  // Stage 1: conversions (8 elems/thread)
  const int CT = 256;
  const int nX8 = (int)(nX / 8), nW8 = (int)(nW / 8);
  cvt_f32_bf16_v8<<<dim3((nX8 + CT - 1) / CT), dim3(CT), 0, stream>>>(q_in, Xq, nX8);
  cvt_f32_bf16_v8<<<dim3((nX8 + CT - 1) / CT), dim3(CT), 0, stream>>>(k_in, Xk, nX8);
  cvt_f32_bf16_v8<<<dim3((nX8 + CT - 1) / CT), dim3(CT), 0, stream>>>(v_in, Xv, nX8);
  cvt_f32_bf16_v8<<<dim3((nW8 + CT - 1) / CT), dim3(CT), 0, stream>>>(Wq, Wqb, nW8);
  cvt_f32_bf16_v8<<<dim3((nW8 + CT - 1) / CT), dim3(CT), 0, stream>>>(Wk, Wkb, nW8);
  cvt_f32_bf16_v8<<<dim3((nW8 + CT - 1) / CT), dim3(CT), 0, stream>>>(Wv, Wvb, nW8);
  cvt_f32_bf16_v8<<<dim3((nW8 + CT - 1) / CT), dim3(CT), 0, stream>>>(Wo, Wob, nW8);

  // Stage 2: projections. Block = 64 rows x 128 cols; grid 4 x 128.
  dim3 pgrid(DMODEL / 128, (BATCH * SEQ) / 64);
  proj_bf16_kernel<<<pgrid, dim3(256), 0, stream>>>(Xq, Wqb, bq, Qb,  0, scale);
  proj_bf16_kernel<<<pgrid, dim3(256), 0, stream>>>(Xk, Wkb, bk, Kb,  0, 1.0f);
  proj_bf16_kernel<<<pgrid, dim3(256), 0, stream>>>(Xv, Wvb, bv, Vtb, 1, 1.0f);

  // Stage 3: fused attention. Grid: (B*H, SEQ/128), 8 waves each own 16 q rows.
  flash_attn_kernel<<<dim3(BATCH * NHEAD, SEQ / 128), dim3(256), 0, stream>>>(
      Qb, Kb, Vtb, Ob);

  // Stage 4: output projection to fp32.
  proj_out_kernel<<<pgrid, dim3(256), 0, stream>>>(Ob, Wob, bo, out);
}
